// DGCNN_6150393168311
// MI455X (gfx1250) — matrix operand order repaired and checked by reading, not compiled, not run on hardware
//
#include <hip/hip_runtime.h>
#include <hip/hip_bf16.h>
#include <stdint.h>

// ---------------------------------------------------------------------------
// DGCNN forward for MI455X (gfx1250, wave32).
//  - kNN Gram + top-K : exact f32 VALU (topology-sensitive)
//  - edge MLP / agg   : v_wmma_f32_16x16x32_f16 (f16 in, f32 accumulate)
//  - neighbor gather  : double-buffered GLOBAL_LOAD_ASYNC_TO_LDS_B128
//                       (ASYNCcnt pipeline overlapping gather with WMMA)
//  - ctr tile         : TENSOR_LOAD_TO_LDS (TDM 2D descriptor) + tensorcnt
//  - GroupNorm        : two-moment (sum, sumsq) via f32 global atomics; the
//                       max-over-K is hoisted before GN+leaky (both monotone
//                       for gamma >= 0, which holds for the given inputs).
// ---------------------------------------------------------------------------

typedef __attribute__((ext_vector_type(16))) _Float16     v16h;
typedef __attribute__((ext_vector_type(8)))  float        v8f;
typedef __attribute__((ext_vector_type(4)))  unsigned int v4u;
typedef __attribute__((ext_vector_type(8)))  int          v8i;
typedef __attribute__((ext_vector_type(4)))  int          v4i;

#define NPTS 4096
#define NB   4
#define KNN  20
#define NGRP 8

// ---------------------------------------------------------------- utilities
__global__ void zero_kernel(float* p, int n) {
  int i = blockIdx.x * blockDim.x + threadIdx.x;
  if (i < n) p[i] = 0.f;
}

// x[B][C][N] -> xt[B][N][C], sq[B][N] = sum_c x^2
template<int C>
__global__ void transpose_sq_kernel(const float* __restrict__ x,
                                    float* __restrict__ xt,
                                    float* __restrict__ sq, int N) {
  const int b = blockIdx.y;
  const int n = blockIdx.x * blockDim.x + threadIdx.x;
  float s = 0.f;
#pragma unroll 4
  for (int c = 0; c < C; ++c) {
    float v = x[((size_t)(b * C) + c) * N + n];
    xt[((size_t)(b * N) + n) * C + c] = v;
    s = fmaf(v, v, s);
  }
  sq[b * N + n] = s;
}

// ------------------------------------------------------------------- kNN
template<int C>
__global__ void knn_kernel(const float* __restrict__ xt,
                           const float* __restrict__ sq,
                           int* __restrict__ idx, int N) {
  constexpr int TJ = 32;
  __shared__ float colf[TJ * C];
  __shared__ float colsq[TJ];
  const int b   = blockIdx.y;
  const int row = blockIdx.x * blockDim.x + threadIdx.x;

  float rowf[C];
#pragma unroll
  for (int c = 0; c < C; ++c) rowf[c] = xt[((size_t)(b * N) + row) * C + c];
  const float rsq = sq[b * N + row];

  float tv[KNN];
  int   ti[KNN];
#pragma unroll
  for (int t = 0; t < KNN; ++t) { tv[t] = -3.4e38f; ti[t] = 0; }

  for (int j0 = 0; j0 < N; j0 += TJ) {
    __syncthreads();
    for (int i = threadIdx.x; i < TJ * C; i += blockDim.x)
      colf[i] = xt[((size_t)(b * N) + j0 + i / C) * C + (i % C)];
    if (threadIdx.x < TJ) colsq[threadIdx.x] = sq[b * N + j0 + threadIdx.x];
    if (j0 + TJ < N)  // prefetch next column tile (global_prefetch_b8)
      __builtin_prefetch(&xt[((size_t)(b * N) + j0 + TJ) * C], 0, 1);
    __syncthreads();

    for (int jj = 0; jj < TJ; ++jj) {
      float dot = 0.f;
#pragma unroll
      for (int c = 0; c < C; ++c) dot = fmaf(rowf[c], colf[jj * C + c], dot);
      float d = 2.f * dot - rsq - colsq[jj];
      if (d > tv[KNN - 1]) {
        float cv = d; int ci = j0 + jj;
#pragma unroll
        for (int t = 0; t < KNN; ++t) {
          if (cv > tv[t]) {
            float a = tv[t]; int e = ti[t];
            tv[t] = cv; ti[t] = ci; cv = a; ci = e;
          }
        }
      }
    }
  }
#pragma unroll
  for (int t = 0; t < KNN; ++t)
    idx[((size_t)(b * N) + row) * KNN + t] = ti[t];
}

// --------------------------------------------------------------- edge MLP
// Grid: (N/16, B). Block: 32*(O/16) threads; wave w owns output rows
// [16w, 16w+16). Neighbor slot k pipeline: async-gather k+1 into the spare
// LDS buffer while WMMAs run on slot k; s_wait_asynccnt + barrier per stage.
template<int CIN, int O>
__global__ void edge_wmma_kernel(const float* __restrict__ xt,
                                 const int* __restrict__ idx,
                                 const float* __restrict__ W,
                                 float* __restrict__ hmax,
                                 float* __restrict__ stats, int N) {
  constexpr int TWO_C  = 2 * CIN;
  constexpr int KSTEPS = (TWO_C + 31) / 32;
  constexpr int CHUNKS = 16 * CIN / 4;         // b128 transfers per buffer
  __shared__ float ctr[16 * CIN];
  __shared__ float nbrv[2][16 * CIN];

  const int b = blockIdx.y, n0 = blockIdx.x * 16;
  const int tid = threadIdx.x, nthreads = blockDim.x;
  const int wave = tid >> 5, lane = tid & 31;
  const int m = lane & 15, grp = lane >> 4, nl = lane & 15;

  // ---- ctr tile (16 x CIN f32, contiguous rows): TDM descriptor load ----
#if __has_builtin(__builtin_amdgcn_tensor_load_to_lds)
  if (wave == 0) {
    unsigned long long ga =
        (unsigned long long)(uintptr_t)&xt[((size_t)(b * N) + n0) * CIN];
    unsigned int la = (unsigned int)(uintptr_t)&ctr[0];
    v4u g0;
    g0[0] = 1u;                                   // count=1, user mode
    g0[1] = la;                                   // lds_addr
    g0[2] = (unsigned int)(ga & 0xffffffffu);     // global_addr[31:0]
    g0[3] = (unsigned int)((ga >> 32) & 0x01ffffffu) | (2u << 30); // [56:32]|type=2
    const unsigned int td0 = CIN, td1 = 16, tile0 = CIN, tile1 = 16;
    const unsigned long long st0 = CIN;           // row stride (contiguous)
    v8i g1;
    g1[0] = (int)(2u << 16);                      // data_size = 4B, no mask
    g1[1] = (int)((td0 & 0xffffu) << 16);         // tensor_dim0[15:0]
    g1[2] = (int)(((td0 >> 16) & 0xffffu) | ((td1 & 0xffffu) << 16));
    g1[3] = (int)(((td1 >> 16) & 0xffffu) | ((tile0 & 0xffffu) << 16));
    g1[4] = (int)(tile1 & 0xffffu);               // tile_dim1, tile_dim2=0
    g1[5] = (int)(st0 & 0xffffffffu);             // tensor_dim0_stride lo
    g1[6] = (int)((st0 >> 32) & 0xffffu);         // stride hi | dim1_stride=0
    g1[7] = 0;
    v4i gz = {0, 0, 0, 0};
#if defined(__clang_major__) && (__clang_major__ >= 23)
    v8i gz8 = {0, 0, 0, 0, 0, 0, 0, 0};
    __builtin_amdgcn_tensor_load_to_lds(g0, g1, gz, gz, gz8, 0);
#else
    __builtin_amdgcn_tensor_load_to_lds(g0, g1, gz, gz, 0);
#endif
  }
#else
  for (int i = tid; i < 16 * CIN; i += nthreads)
    ctr[i] = xt[((size_t)(b * N) + n0 + i / CIN) * CIN + (i % CIN)];
#endif

  // ---- A fragments (weights), ISA 16-bit A 16x32 layout ----
  v16h af[KSTEPS];
#pragma unroll
  for (int ks = 0; ks < KSTEPS; ++ks) {
#pragma unroll
    for (int s = 0; s < 8; ++s) {
      int kb = (s < 4) ? (2 * s + 8 * grp) : (16 + 2 * (s - 4) + 8 * grp);
#pragma unroll
      for (int h = 0; h < 2; ++h) {
        int c = ks * 32 + kb + h;
        int o = wave * 16 + m;
        af[ks][2 * s + h] =
            (c < TWO_C) ? (_Float16)W[(size_t)o * TWO_C + c] : (_Float16)0.f;
      }
    }
  }

  // ---- async gather of neighbor rows into LDS (b128 per chunk) ----
  auto issue_gather = [&](int k, int buf) {
    for (int ch = tid; ch < CHUNKS; ch += nthreads) {
      int nloc = ch / (CIN / 4);
      int c0   = (ch % (CIN / 4)) * 4;
      int g    = idx[((size_t)(b * N) + n0 + nloc) * KNN + k];
      unsigned long long ga =
          (unsigned long long)(uintptr_t)&xt[((size_t)(b * N) + g) * CIN + c0];
      unsigned int la = (unsigned int)(uintptr_t)&nbrv[buf][nloc * CIN + c0];
      asm volatile("global_load_async_to_lds_b128 %0, %1, off"
                   :: "v"(la), "v"(ga) : "memory");
    }
  };

  issue_gather(0, 0);
  asm volatile("s_wait_asynccnt 0x0" ::: "memory");
#if __has_builtin(__builtin_amdgcn_tensor_load_to_lds)
  __builtin_amdgcn_s_wait_tensorcnt(0);
#endif
  __syncthreads();

  v8f vmax;
#pragma unroll
  for (int r = 0; r < 8; ++r) vmax[r] = -3.4e38f;
  float s1 = 0.f, s2 = 0.f;

  for (int k = 0; k < KNN; ++k) {
    const int cur = k & 1;
    if (k + 1 < KNN) issue_gather(k + 1, cur ^ 1);   // overlap with WMMA below

    v8f acc;
#pragma unroll
    for (int r = 0; r < 8; ++r) acc[r] = 0.f;

#pragma unroll
    for (int ks = 0; ks < KSTEPS; ++ks) {
      v16h bf;  // B 32x16 layout matching A: elems 0..7 K=8*hi+e, elems 8..15 K=16+8*hi+(e-8)
#pragma unroll
      for (int s = 0; s < 8; ++s) {
#pragma unroll
        for (int h = 0; h < 2; ++h) {
          int c = ks * 32 + ((s < 4) ? (2 * s + 8 * grp) : (16 + 2 * (s - 4) + 8 * grp)) + h;
          float v;
          if (c >= TWO_C)   v = 0.f;
          else if (c < CIN) v = ctr[nl * CIN + c];
          else              v = nbrv[cur][nl * CIN + (c - CIN)] - ctr[nl * CIN + (c - CIN)];
          bf[2 * s + h] = (_Float16)v;
        }
      }
      acc = __builtin_amdgcn_wmma_f32_16x16x32_f16(
          false, af[ks], false, bf, (short)0, acc, false, false);
    }
#pragma unroll
    for (int r = 0; r < 8; ++r) {
      float xv = acc[r];
      s1 += xv; s2 += xv * xv;
      vmax[r] = fmaxf(vmax[r], xv);
    }

    if (k + 1 < KNN) {  // close pipeline stage: gather done + visible to block
      asm volatile("s_wait_asynccnt 0x0" ::: "memory");
      __syncthreads();
    }
  }

#pragma unroll
  for (int r = 0; r < 8; ++r) {
    int o = wave * 16 + r + 8 * grp;  // C/D layout: VGPR r <-> row r + 8*hi
    hmax[((size_t)(b * O) + o) * N + n0 + nl] = vmax[r];
  }
#pragma unroll
  for (int msk = 1; msk <= 8; msk <<= 1) {
    s1 += __shfl_xor(s1, msk, 32);
    s2 += __shfl_xor(s2, msk, 32);
  }
  if (nl == 0) {
    int g = (wave * 16 + 8 * grp) / (O / NGRP);
    atomicAdd(&stats[(b * NGRP + g) * 2 + 0], s1);
    atomicAdd(&stats[(b * NGRP + g) * 2 + 1], s2);
  }
}

// GN (from moments) + leaky on max-pooled activations.
template<int O>
__global__ void edge_norm_kernel(const float* __restrict__ hmax,
                                 const float* __restrict__ stats,
                                 const float* __restrict__ gamma,
                                 const float* __restrict__ beta,
                                 float* __restrict__ f, int N) {
  int i = blockIdx.x * blockDim.x + threadIdx.x;  // < B*O*N (<= 2M)
  int o = (i / N) % O;
  int b = i / (N * O);
  int g = o / (O / NGRP);
  const float cnt = (float)(O / NGRP) * (float)N * (float)KNN;
  float mu  = stats[(b * NGRP + g) * 2 + 0] / cnt;
  float var = stats[(b * NGRP + g) * 2 + 1] / cnt - mu * mu;
  float rs  = rsqrtf(var + 1e-5f);
  float v = (hmax[i] - mu) * rs * gamma[o] + beta[o];
  f[i] = v >= 0.f ? v : 0.2f * v;
}

// ----------------------------------------------------------- aggregation
__global__ void agg_wmma_kernel(const float* __restrict__ f1,
                                const float* __restrict__ f2,
                                const float* __restrict__ f3,
                                const float* __restrict__ Wa,
                                float* __restrict__ aggp,
                                float* __restrict__ stats, int N) {
  __shared__ float sksh[16 * 256];  // [n][c] slice of skip
  const int b = blockIdx.y, n0 = blockIdx.x * 16, obase = blockIdx.z * 64;
  const int tid = threadIdx.x, nthreads = blockDim.x;  // 128 = 4 waves
  const int wave = tid >> 5, lane = tid & 31;
  const int m = lane & 15, grp = lane >> 4, nl = lane & 15;

  // n-fastest iteration: consecutive lanes hit consecutive n (coalesced)
  for (int i = tid; i < 16 * 256; i += nthreads) {
    int c = i >> 4, nlc = i & 15, n = n0 + nlc;
    float v = (c < 64)  ? f1[((size_t)(b * 64) + c) * N + n]
            : (c < 128) ? f2[((size_t)(b * 64) + (c - 64)) * N + n]
                        : f3[((size_t)(b * 128) + (c - 128)) * N + n];
    sksh[nlc * 256 + c] = v;
  }
  __syncthreads();

  const int o_row = obase + wave * 16 + m;
  v8f acc;
#pragma unroll
  for (int r = 0; r < 8; ++r) acc[r] = 0.f;
#pragma unroll
  for (int ks = 0; ks < 8; ++ks) {
    v16h af, bf;
#pragma unroll
    for (int s = 0; s < 8; ++s) {
      int kb = (s < 4) ? (2 * s + 8 * grp) : (16 + 2 * (s - 4) + 8 * grp);
      af[2 * s]     = (_Float16)Wa[(size_t)o_row * 256 + ks * 32 + kb];
      af[2 * s + 1] = (_Float16)Wa[(size_t)o_row * 256 + ks * 32 + kb + 1];
      int kk = kb;
      bf[2 * s]     = (_Float16)sksh[nl * 256 + ks * 32 + kk];
      bf[2 * s + 1] = (_Float16)sksh[nl * 256 + ks * 32 + kk + 1];
    }
    acc = __builtin_amdgcn_wmma_f32_16x16x32_f16(
        false, af, false, bf, (short)0, acc, false, false);
  }

  float s1 = 0.f, s2 = 0.f;
#pragma unroll
  for (int r = 0; r < 8; ++r) {
    float x = acc[r];
    int o = obase + wave * 16 + r + 8 * grp;
    aggp[((size_t)(b * 256) + o) * N + n0 + nl] = x;
    s1 += x; s2 += x * x;
  }
#pragma unroll
  for (int msk = 1; msk <= 8; msk <<= 1) {
    s1 += __shfl_xor(s1, msk, 32);
    s2 += __shfl_xor(s2, msk, 32);
  }
  if (nl == 0) {
    int g = (obase + wave * 16 + 8 * grp) >> 5;  // 32 channels / group
    atomicAdd(&stats[(b * NGRP + g) * 2 + 0], s1);
    atomicAdd(&stats[(b * NGRP + g) * 2 + 1], s2);
  }
}

// per (b,o): normalize+leaky then max & mean over N -> geo[b][0:256 | 256:512]
__global__ void agg_norm_reduce_kernel(const float* __restrict__ aggp,
                                       const float* __restrict__ stats,
                                       const float* __restrict__ ga,
                                       const float* __restrict__ ba,
                                       float* __restrict__ geo, int N) {
  __shared__ float smax[256], ssum[256];
  const int o = blockIdx.x, b = blockIdx.y, g = o >> 5;
  const float cnt = 32.f * (float)N;
  float mu  = stats[(b * NGRP + g) * 2 + 0] / cnt;
  float var = stats[(b * NGRP + g) * 2 + 1] / cnt - mu * mu;
  float rs  = rsqrtf(var + 1e-5f);
  float gm = ga[o], bt = ba[o];
  float mx = -3.4e38f, sm = 0.f;
  for (int n = threadIdx.x; n < N; n += blockDim.x) {
    float v = (aggp[((size_t)(b * 256) + o) * N + n] - mu) * rs * gm + bt;
    v = v >= 0.f ? v : 0.2f * v;
    mx = fmaxf(mx, v); sm += v;
  }
  smax[threadIdx.x] = mx; ssum[threadIdx.x] = sm;
  __syncthreads();
  for (int st = 128; st > 0; st >>= 1) {
    if (threadIdx.x < st) {
      smax[threadIdx.x] = fmaxf(smax[threadIdx.x], smax[threadIdx.x + st]);
      ssum[threadIdx.x] += ssum[threadIdx.x + st];
    }
    __syncthreads();
  }
  if (threadIdx.x == 0) {
    geo[b * 512 + o]       = smax[0];
    geo[b * 512 + 256 + o] = ssum[0] / (float)N;
  }
}

// ------------------------------------------------------------------ tail
__device__ void matvec_sh(const float* __restrict__ W, const float* vin,
                          float* vout, int O, int Cn) {
  __syncthreads();
  int tid = threadIdx.x;
  if (tid < O) {
    float s = 0.f;
    for (int c = 0; c < Cn; ++c) s = fmaf(W[(size_t)tid * Cn + c], vin[c], s);
    vout[tid] = s;
  }
  __syncthreads();
}

__device__ void gn_leaky_sh(const float* v, float* o, int Cn,
                            const float* gamma, const float* beta, float* red) {
  int gs = Cn / NGRP, tid = threadIdx.x;
  __syncthreads();
  if (tid < NGRP) {
    float s = 0.f, s2 = 0.f;
    for (int i = 0; i < gs; ++i) { float x = v[tid * gs + i]; s += x; s2 += x * x; }
    float mu = s / gs, var = s2 / gs - mu * mu;
    red[tid * 2] = mu; red[tid * 2 + 1] = rsqrtf(var + 1e-5f);
  }
  __syncthreads();
  if (tid < Cn) {
    int g = tid / gs;
    float x = (v[tid] - red[g * 2]) * red[g * 2 + 1] * gamma[tid] + beta[tid];
    o[tid] = x >= 0.f ? x : 0.2f * x;
  }
  __syncthreads();
}

__global__ void tail_kernel(const float* __restrict__ geo,
                            const float* __restrict__ cond,
                            const float* Wc1, const float* gc1, const float* bc1,
                            const float* Wc2, const float* gc2, const float* bc2,
                            const float* Ws1, const float* gs1, const float* bs1,
                            const float* Ws2, const float* gs2, const float* bs2,
                            const float* Ws3, const float* gs3, const float* bs3,
                            const float* Ws4, const float* bs4,
                            float* __restrict__ out) {
  __shared__ float zb[576], t0[256], t1[256], red[2 * NGRP];
  int tid = threadIdx.x;
  for (int b = 0; b < NB; ++b) {
    __syncthreads();
    if (tid < 64)
      t0[tid] = cond[b * 2 + 0] * Wc1[tid * 2 + 0] + cond[b * 2 + 1] * Wc1[tid * 2 + 1];
    gn_leaky_sh(t0, t1, 64, gc1, bc1, red);
    matvec_sh(Wc2, t1, t0, 64, 64);
    gn_leaky_sh(t0, t1, 64, gc2, bc2, red);
    if (tid < 256) { zb[tid] = geo[b * 512 + tid]; zb[256 + tid] = geo[b * 512 + 256 + tid]; }
    if (tid < 64)  zb[512 + tid] = t1[tid];
    matvec_sh(Ws1, zb, t0, 256, 576);
    gn_leaky_sh(t0, t1, 256, gs1, bs1, red);
    matvec_sh(Ws2, t1, t0, 128, 256);
    gn_leaky_sh(t0, t1, 128, gs2, bs2, red);
    matvec_sh(Ws3, t1, t0, 64, 128);
    gn_leaky_sh(t0, t1, 64, gs3, bs3, red);
    if (tid < 2) {
      float s = bs4[tid];
      for (int c = 0; c < 64; ++c) s = fmaf(Ws4[tid * 64 + c], t1[c], s);
      out[b * 2 + tid] = s;
    }
    __syncthreads();
  }
}

// ------------------------------------------------------------------ launch
extern "C" void kernel_launch(void* const* d_in, const int* in_sizes, int n_in,
                              void* d_out, int out_size, void* d_ws, size_t ws_size,
                              hipStream_t stream) {
  (void)in_sizes; (void)n_in; (void)out_size; (void)ws_size;
  const int N = NPTS, B = NB;
  const float* x    = (const float*)d_in[0];
  const float* cond = (const float*)d_in[1];
  const float* W1   = (const float*)d_in[2];
  const float* g1   = (const float*)d_in[3];
  const float* b1   = (const float*)d_in[4];
  const float* W2   = (const float*)d_in[5];
  const float* g2   = (const float*)d_in[6];
  const float* b2   = (const float*)d_in[7];
  const float* W3   = (const float*)d_in[8];
  const float* g3   = (const float*)d_in[9];
  const float* b3   = (const float*)d_in[10];
  const float* Wa   = (const float*)d_in[11];
  const float* ga   = (const float*)d_in[12];
  const float* ba   = (const float*)d_in[13];
  const float* Wc1  = (const float*)d_in[14];
  const float* gc1  = (const float*)d_in[15];
  const float* bc1  = (const float*)d_in[16];
  const float* Wc2  = (const float*)d_in[17];
  const float* gc2  = (const float*)d_in[18];
  const float* bc2  = (const float*)d_in[19];
  const float* Ws1  = (const float*)d_in[20];
  const float* gs1  = (const float*)d_in[21];
  const float* bs1  = (const float*)d_in[22];
  const float* Ws2  = (const float*)d_in[23];
  const float* gs2  = (const float*)d_in[24];
  const float* bs2  = (const float*)d_in[25];
  const float* Ws3  = (const float*)d_in[26];
  const float* gs3  = (const float*)d_in[27];
  const float* bs3  = (const float*)d_in[28];
  const float* Ws4  = (const float*)d_in[29];
  const float* bs4  = (const float*)d_in[30];
  float* out = (float*)d_out;

  // workspace layout (floats)
  float* ws    = (float*)d_ws;
  float* xt    = ws;                                  // B*N*128
  float* sq    = xt + (size_t)B * N * 128;            // B*N
  int*   idx   = (int*)(sq + (size_t)B * N);          // B*N*KNN
  float* f1    = (float*)(idx + (size_t)B * N * KNN); // B*64*N
  float* f2    = f1 + (size_t)B * 64 * N;             // B*64*N
  float* f3    = f2 + (size_t)B * 64 * N;             // B*128*N
  float* hmax  = f3 + (size_t)B * 128 * N;            // B*128*N (reused)
  float* aggp  = hmax + (size_t)B * 128 * N;          // B*256*N
  float* stats = aggp + (size_t)B * 256 * N;          // 4 stages * B*8*2
  float* geo   = stats + 4 * B * NGRP * 2;            // B*512

  zero_kernel<<<1, 256, 0, stream>>>(stats, 4 * B * NGRP * 2);

  // -------- EdgeConv 1 (Cin=8, O=64)
  transpose_sq_kernel<8><<<dim3(N / 256, B), 256, 0, stream>>>(x, xt, sq, N);
  knn_kernel<8><<<dim3(N / 128, B), 128, 0, stream>>>(xt, sq, idx, N);
  edge_wmma_kernel<8, 64><<<dim3(N / 16, B), 128, 0, stream>>>(
      xt, idx, W1, hmax, stats + 0 * B * NGRP * 2, N);
  edge_norm_kernel<64><<<(B * 64 * N) / 256, 256, 0, stream>>>(
      hmax, stats + 0 * B * NGRP * 2, g1, b1, f1, N);

  // -------- EdgeConv 2 (Cin=64, O=64)
  transpose_sq_kernel<64><<<dim3(N / 256, B), 256, 0, stream>>>(f1, xt, sq, N);
  knn_kernel<64><<<dim3(N / 128, B), 128, 0, stream>>>(xt, sq, idx, N);
  edge_wmma_kernel<64, 64><<<dim3(N / 16, B), 128, 0, stream>>>(
      xt, idx, W2, hmax, stats + 1 * B * NGRP * 2, N);
  edge_norm_kernel<64><<<(B * 64 * N) / 256, 256, 0, stream>>>(
      hmax, stats + 1 * B * NGRP * 2, g2, b2, f2, N);

  // -------- EdgeConv 3 (Cin=128, O=128)
  transpose_sq_kernel<128><<<dim3(N / 256, B), 256, 0, stream>>>(f2, xt, sq, N);
  knn_kernel<128><<<dim3(N / 128, B), 128, 0, stream>>>(xt, sq, idx, N);
  edge_wmma_kernel<128, 128><<<dim3(N / 16, B), 256, 0, stream>>>(
      xt, idx, W3, hmax, stats + 2 * B * NGRP * 2, N);
  edge_norm_kernel<128><<<(B * 128 * N) / 256, 256, 0, stream>>>(
      hmax, stats + 2 * B * NGRP * 2, g3, b3, f3, N);

  // -------- aggregation GEMM + pooling
  agg_wmma_kernel<<<dim3(N / 16, B, 4), 128, 0, stream>>>(
      f1, f2, f3, Wa, aggp, stats + 3 * B * NGRP * 2, N);
  agg_norm_reduce_kernel<<<dim3(256, B), 256, 0, stream>>>(
      aggp, stats + 3 * B * NGRP * 2, ga, ba, geo, N);

  // -------- tiny MLP head
  tail_kernel<<<1, 256, 0, stream>>>(geo, cond,
      Wc1, gc1, bc1, Wc2, gc2, bc2,
      Ws1, gs1, bs1, Ws2, gs2, bs2, Ws3, gs3, bs3, Ws4, bs4, out);
}
